// LoFTREncoderLayer_61426622267547
// MI455X (gfx1250) — matrix operand order repaired and checked by reading, not compiled, not run on hardware
//
#include <hip/hip_runtime.h>
#include <hip/hip_bf16.h>

typedef _Float16 half_t;
typedef __attribute__((ext_vector_type(8)))  _Float16 v8h;
typedef __attribute__((ext_vector_type(16))) _Float16 v16h;
typedef __attribute__((ext_vector_type(8)))  float    v8f;
typedef int async_i4 __attribute__((vector_size(16)));   // matches builtin's V4i param

#define PITCH 48          // LDS row pitch in halfs (96B, 16B-aligned)
#define Lq    4800
#define Lp    4864        // L padded to 19*256 for the msg kernel
#define CC    256
#define HH    8
#define DD    32
#define MR    19200       // N*L

// ---- CDNA5 async global->LDS path (guarded; falls back to VGPR staging) ----
#ifndef HAVE_ASYNC_LDS
#if defined(__has_builtin)
#if __has_builtin(__builtin_amdgcn_global_load_async_to_lds_b128)
#define HAVE_ASYNC_LDS 1
#endif
#endif
#endif
#ifndef HAVE_ASYNC_LDS
#define HAVE_ASYNC_LDS 0
#endif

__device__ __forceinline__ void stage_fence() {
#if HAVE_ASYNC_LDS
#if defined(__has_builtin) && __has_builtin(__builtin_amdgcn_s_wait_asynccnt)
  __builtin_amdgcn_s_wait_asynccnt(0);
#else
  asm volatile("s_wait_asynccnt 0" ::: "memory");
#endif
#endif
  __syncthreads();
}

__device__ __forceinline__ v8f wmma32(v16h a, v16h b, v8f c) {
  return __builtin_amdgcn_wmma_f32_16x16x32_f16(false, a, false, b, (short)0, c, false, false);
}

// ---- fragment loaders (16-bit A 16x32: lanes 0-15 rows, K = g*16..g*16+15 contiguous) ----
__device__ __forceinline__ v16h frag_a_pitch(const half_t* base, int pitch) {
  int lane = threadIdx.x & 31;
  int g = lane >> 4, r = lane & 15;
  v8h lo = *(const v8h*)(base + (size_t)r * pitch + g * 16);
  v8h hi = *(const v8h*)(base + (size_t)r * pitch + g * 16 + 8);
  v16h f;
#pragma unroll
  for (int i = 0; i < 8; ++i) { f[i] = lo[i]; f[i + 8] = hi[i]; }
  return f;
}
// B 32x16, column-major rows (n-major): lane n=l&15, K = g*16..g*16+15 contiguous
__device__ __forceinline__ v16h frag_b_pitch(const half_t* base, int pitch) {
  int lane = threadIdx.x & 31;
  int g = lane >> 4, n = lane & 15;
  const v8h* p = (const v8h*)(base + (size_t)n * pitch + g * 16);
  v8h lo = p[0], hi = p[1];
  v16h f;
#pragma unroll
  for (int i = 0; i < 8; ++i) { f[i] = lo[i]; f[i + 8] = hi[i]; }
  return f;
}

// stage rows x 32-half tile, row-major global -> LDS (pitch 48), 256 threads.
// Uses CDNA5 async-to-LDS b128 transfers when available (ASYNCcnt tracked).
__device__ __forceinline__ void stage_tile(const half_t* g, int ldg, half_t* lds,
                                           int rows, int r0, int k0) {
  int total = rows * 4;
  for (int c = threadIdx.x; c < total; c += 256) {
    int r = c >> 2, ch = c & 3;
    const half_t* src = g + (size_t)(r0 + r) * ldg + k0 + ch * 8;
    half_t* dst = lds + r * PITCH + ch * 8;
#if HAVE_ASYNC_LDS
    __builtin_amdgcn_global_load_async_to_lds_b128(
        (__attribute__((address_space(1))) async_i4*)src,
        (__attribute__((address_space(3))) async_i4*)dst, 0, 0);
#else
    *(float4*)dst = *(const float4*)src;
#endif
  }
}

__global__ void cvt_f16(const float* __restrict__ in, half_t* __restrict__ out, int n) {
  int i = blockIdx.x * 256 + threadIdx.x;
  if (i < n) out[i] = (half_t)in[i];
}

// ---------------- QKV projection + rotary + phi, 128x64 tiles, K=256 ----------------
// MODE 0: Q -> Qphi [N,H,Lp,D];  MODE 1: K -> KphiT [N,H,D,S];  MODE 2: V/S -> VscT [N,H,D,S]
template <int MODE>
__global__ __launch_bounds__(256) void proj_kernel(const half_t* __restrict__ A,
                                                   const half_t* __restrict__ W,
                                                   const float* __restrict__ pe,
                                                   half_t* __restrict__ out) {
  __shared__ __align__(16) half_t ldsA[2][128 * PITCH];
  __shared__ __align__(16) half_t ldsB[2][64 * PITCH];
  int m0 = blockIdx.x * 128, n0 = blockIdx.y * 64;
  int wid = threadIdx.x >> 5, wm = wid >> 1, wn = wid & 1;
  v8f acc[2][2] = {};
  stage_tile(A, 256, ldsA[0], 128, m0, 0);
  stage_tile(W, 256, ldsB[0], 64, n0, 0);
  stage_fence();
  for (int kk = 0; kk < 256; kk += 32) {
    int cur = (kk >> 5) & 1;
    if (kk + 32 < 256) {           // prefetch next K-tile into the other buffer
      stage_tile(A, 256, ldsA[1 - cur], 128, m0, kk + 32);
      stage_tile(W, 256, ldsB[1 - cur], 64, n0, kk + 32);
    }
    v16h a0 = frag_a_pitch(ldsA[cur] + (wm * 32) * PITCH, PITCH);
    v16h a1 = frag_a_pitch(ldsA[cur] + (wm * 32 + 16) * PITCH, PITCH);
    v16h b0 = frag_b_pitch(ldsB[cur] + (wn * 32) * PITCH, PITCH);
    v16h b1 = frag_b_pitch(ldsB[cur] + (wn * 32 + 16) * PITCH, PITCH);
    acc[0][0] = wmma32(a0, b0, acc[0][0]);
    acc[0][1] = wmma32(a0, b1, acc[0][1]);
    acc[1][0] = wmma32(a1, b0, acc[1][0]);
    acc[1][1] = wmma32(a1, b1, acc[1][1]);
    stage_fence();
  }
  int lane = threadIdx.x & 31, nloc = lane & 15, rbase = (lane >> 4) * 8;
#pragma unroll
  for (int mf = 0; mf < 2; ++mf)
#pragma unroll
    for (int nf = 0; nf < 2; ++nf)
#pragma unroll
      for (int i = 0; i < 8; ++i) {
        float t = acc[mf][nf][i];
        int row = m0 + wm * 32 + mf * 16 + rbase + i;
        int c = n0 + wn * 32 + nf * 16 + nloc;
        int nb = row / Lq, l = row - nb * Lq;
        int h = c >> 5, d = c & 31;
        if (MODE <= 1) {
          float partner = __shfl_xor(t, 1, 32);  // pair channel lives in adjacent lane
          float2 cs = ((const float2*)pe)[(size_t)row * CC + c];
          float t2 = (c & 1) ? partner : -partner;
          float r = t * cs.x + t2 * cs.y;
          float phi = (r > 0.f) ? (r + 1.f) : __expf(r);   // elu(r)+1
          if (MODE == 0)
            out[((size_t)(nb * HH + h) * Lp + l) * DD + d] = (half_t)phi;
          else
            out[((size_t)(nb * HH + h) * DD + d) * Lq + l] = (half_t)phi;
        } else {
          out[((size_t)(nb * HH + h) * DD + d) * Lq + l] = (half_t)(t * (1.f / Lq));
        }
      }
}

// ---------------- Ksum[n,h,d] = sum_s KphiT[n,h,d,s] ----------------
__global__ __launch_bounds__(256) void ksum_kernel(const half_t* __restrict__ KphiT,
                                                   float* __restrict__ Ksum) {
  __shared__ float red[256];
  int row = blockIdx.x;
  const half_t* p = KphiT + (size_t)row * Lq;
  float s = 0.f;
  for (int i = threadIdx.x; i < Lq; i += 256) s += (float)p[i];
  red[threadIdx.x] = s;
  __syncthreads();
  for (int off = 128; off; off >>= 1) {
    if ((int)threadIdx.x < off) red[threadIdx.x] += red[threadIdx.x + off];
    __syncthreads();
  }
  if (threadIdx.x == 0) Ksum[row] = red[0];
}

// ---------------- KV[n,h,d,e] = sum_s Kphi[s,d]*Vsc[s,e] : 32x32 per (n,h) ----------------
__global__ __launch_bounds__(256) void kv_kernel(const half_t* __restrict__ KphiT,
                                                 const half_t* __restrict__ VscT,
                                                 float* __restrict__ KV) {
  int nh = blockIdx.x;
  const half_t* K = KphiT + (size_t)nh * DD * Lq;
  const half_t* V = VscT + (size_t)nh * DD * Lq;
  int wid = threadIdx.x >> 5;
  v8f acc[2][2] = {};
  for (int c = wid; c < Lq / 32; c += 8) {
    int s0 = c * 32;
    v16h a0 = frag_a_pitch(K + s0, Lq);
    v16h a1 = frag_a_pitch(K + (size_t)16 * Lq + s0, Lq);
    v16h b0 = frag_b_pitch(V + s0, Lq);
    v16h b1 = frag_b_pitch(V + (size_t)16 * Lq + s0, Lq);
    acc[0][0] = wmma32(a0, b0, acc[0][0]);
    acc[0][1] = wmma32(a0, b1, acc[0][1]);
    acc[1][0] = wmma32(a1, b0, acc[1][0]);
    acc[1][1] = wmma32(a1, b1, acc[1][1]);
  }
  __shared__ float red[8][1024];
  int lane = threadIdx.x & 31, nloc = lane & 15, rbase = (lane >> 4) * 8;
#pragma unroll
  for (int mf = 0; mf < 2; ++mf)
#pragma unroll
    for (int nf = 0; nf < 2; ++nf)
#pragma unroll
      for (int i = 0; i < 8; ++i) {
        int d = mf * 16 + rbase + i, e = nf * 16 + nloc;
        red[wid][d * 32 + e] = acc[mf][nf][i];
      }
  __syncthreads();
  for (int idx = threadIdx.x; idx < 1024; idx += 256) {
    float s = 0.f;
#pragma unroll
    for (int w = 0; w < 8; ++w) s += red[w][idx];   // fixed order -> deterministic
    KV[(size_t)nh * 1024 + idx] = s;
  }
}

// ---------------- msg[l, h*32+e] = Z(l) * (Qphi[l,:] @ KV) ----------------
__global__ __launch_bounds__(256) void msg_kernel(const half_t* __restrict__ Qphi,
                                                  const float* __restrict__ KV,
                                                  const float* __restrict__ Ksum,
                                                  half_t* __restrict__ msg) {
  int lt = blockIdx.x, nh = blockIdx.y;
  __shared__ __align__(16) half_t bt[32 * PITCH];
  __shared__ float ks[32];
  __shared__ float zs[256];
  const float* kv = KV + (size_t)nh * 1024;
  for (int idx = threadIdx.x; idx < 1024; idx += 256) {
    int d = idx >> 5, e = idx & 31;
    bt[e * PITCH + d] = (half_t)kv[idx];   // transpose: B rows are columns e
  }
  if (threadIdx.x < 32) ks[threadIdx.x] = Ksum[nh * 32 + threadIdx.x];
  __syncthreads();
  const half_t* Q = Qphi + (size_t)nh * Lp * DD;
  {
    int l = lt * 256 + threadIdx.x;
    const half_t* q = Q + (size_t)l * DD;
    float s = 0.f;
#pragma unroll
    for (int d = 0; d < 32; ++d) s += (float)q[d] * ks[d];
    zs[threadIdx.x] = (float)Lq / (s + 1e-6f);
  }
  __syncthreads();
  int wid = threadIdx.x >> 5;
  int l0 = lt * 256 + wid * 32;
  v16h a0 = frag_a_pitch(Q + (size_t)l0 * DD, DD);
  v16h a1 = frag_a_pitch(Q + (size_t)(l0 + 16) * DD, DD);
  v16h b0 = frag_b_pitch(bt, PITCH);
  v16h b1 = frag_b_pitch(bt + 16 * PITCH, PITCH);
  v8f acc[2][2] = {};
  acc[0][0] = wmma32(a0, b0, acc[0][0]);
  acc[0][1] = wmma32(a0, b1, acc[0][1]);
  acc[1][0] = wmma32(a1, b0, acc[1][0]);
  acc[1][1] = wmma32(a1, b1, acc[1][1]);
  int lane = threadIdx.x & 31, nloc = lane & 15, rbase = (lane >> 4) * 8;
  int nb = nh >> 3, h = nh & 7;
#pragma unroll
  for (int mf = 0; mf < 2; ++mf)
#pragma unroll
    for (int nf = 0; nf < 2; ++nf)
#pragma unroll
      for (int i = 0; i < 8; ++i) {
        int lr = wid * 32 + mf * 16 + rbase + i;
        int lg = lt * 256 + lr;
        if (lg < Lq) {
          int e = nf * 16 + nloc;
          float v = acc[mf][nf][i] * zs[lr];
          msg[((size_t)(nb * Lq + lg)) * CC + h * 32 + e] = (half_t)v;
        }
      }
}

// ---------------- generic 64x256 GEMM + LayerNorm epilogues ----------------
// MODE 0: merge: (msg @ Wm^T), LN(g1,b1) -> cat[:,256:512]; cat[:,0:256] = xh
// MODE 1: mlp2 : (relu @ W2^T), LN(g2,b2) + x residual -> out (f32)
template <int MODE, int KTOT>
__global__ __launch_bounds__(256) void gemm_ln_kernel(const half_t* __restrict__ A,
                                                      const half_t* __restrict__ W,
                                                      const half_t* __restrict__ xh,
                                                      const float* __restrict__ x,
                                                      const float* __restrict__ gg,
                                                      const float* __restrict__ bb,
                                                      half_t* __restrict__ cat,
                                                      float* __restrict__ out) {
  __shared__ __align__(16) half_t ldsA[2][64 * PITCH];
  __shared__ __align__(16) half_t ldsB[2][256 * PITCH];
  __shared__ float part[4][64][2];
  int m0 = blockIdx.x * 64;
  int wid = threadIdx.x >> 5, wm = wid >> 2, wn = wid & 3;
  v8f acc[2][4] = {};
  stage_tile(A, KTOT, ldsA[0], 64, m0, 0);
  stage_tile(W, KTOT, ldsB[0], 256, 0, 0);
  stage_fence();
  for (int kk = 0; kk < KTOT; kk += 32) {
    int cur = (kk >> 5) & 1;
    if (kk + 32 < KTOT) {
      stage_tile(A, KTOT, ldsA[1 - cur], 64, m0, kk + 32);
      stage_tile(W, KTOT, ldsB[1 - cur], 256, 0, kk + 32);
    }
    v16h a0 = frag_a_pitch(ldsA[cur] + (wm * 32) * PITCH, PITCH);
    v16h a1 = frag_a_pitch(ldsA[cur] + (wm * 32 + 16) * PITCH, PITCH);
#pragma unroll
    for (int nf = 0; nf < 4; ++nf) {
      v16h b = frag_b_pitch(ldsB[cur] + (wn * 64 + nf * 16) * PITCH, PITCH);
      acc[0][nf] = wmma32(a0, b, acc[0][nf]);
      acc[1][nf] = wmma32(a1, b, acc[1][nf]);
    }
    stage_fence();
  }
  int lane = threadIdx.x & 31, nloc = lane & 15, rbase = (lane >> 4) * 8;
  // per-row partial mean / meansq, 16-lane shuffle reduce, cross-wave via LDS
#pragma unroll
  for (int mf = 0; mf < 2; ++mf)
#pragma unroll
    for (int i = 0; i < 8; ++i) {
      float s = 0.f, ss = 0.f;
#pragma unroll
      for (int nf = 0; nf < 4; ++nf) {
        float v = acc[mf][nf][i];
        s += v; ss += v * v;
      }
#pragma unroll
      for (int off = 1; off < 16; off <<= 1) {
        s += __shfl_xor(s, off, 32);
        ss += __shfl_xor(ss, off, 32);
      }
      int lr = wm * 32 + mf * 16 + rbase + i;
      if (nloc == 0) { part[wn][lr][0] = s; part[wn][lr][1] = ss; }
    }
  __syncthreads();
#pragma unroll
  for (int mf = 0; mf < 2; ++mf)
#pragma unroll
    for (int i = 0; i < 8; ++i) {
      int lr = wm * 32 + mf * 16 + rbase + i;
      float s = part[0][lr][0] + part[1][lr][0] + part[2][lr][0] + part[3][lr][0];
      float ss = part[0][lr][1] + part[1][lr][1] + part[2][lr][1] + part[3][lr][1];
      float mu = s * (1.f / 256.f);
      float var = ss * (1.f / 256.f) - mu * mu;
      float rs = rsqrtf(var + 1e-5f);
      size_t row = m0 + lr;
#pragma unroll
      for (int nf = 0; nf < 4; ++nf) {
        int c = wn * 64 + nf * 16 + nloc;
        float v = (acc[mf][nf][i] - mu) * rs * gg[c] + bb[c];
        if (MODE == 0)
          cat[row * 512 + 256 + c] = (half_t)v;
        else
          out[row * 256 + c] = x[row * 256 + c] + v;
      }
    }
  if (MODE == 0) {  // cat[:,0:256] = x (already f16 in xh)
    for (int idx = threadIdx.x; idx < 64 * 256; idx += 256) {
      int r = idx >> 8, c = idx & 255;
      cat[(size_t)(m0 + r) * 512 + c] = xh[(size_t)(m0 + r) * 256 + c];
    }
  }
}

// ---------------- MLP1: cat[M,512] @ W1^T -> relu, 128x64 tiles ----------------
__global__ __launch_bounds__(256) void mlp1_kernel(const half_t* __restrict__ A,
                                                   const half_t* __restrict__ W,
                                                   half_t* __restrict__ relu) {
  __shared__ __align__(16) half_t ldsA[2][128 * PITCH];
  __shared__ __align__(16) half_t ldsB[2][64 * PITCH];
  int m0 = blockIdx.x * 128, n0 = blockIdx.y * 64;
  int wid = threadIdx.x >> 5, wm = wid >> 1, wn = wid & 1;
  v8f acc[2][2] = {};
  stage_tile(A, 512, ldsA[0], 128, m0, 0);
  stage_tile(W, 512, ldsB[0], 64, n0, 0);
  stage_fence();
  for (int kk = 0; kk < 512; kk += 32) {
    int cur = (kk >> 5) & 1;
    if (kk + 32 < 512) {
      stage_tile(A, 512, ldsA[1 - cur], 128, m0, kk + 32);
      stage_tile(W, 512, ldsB[1 - cur], 64, n0, kk + 32);
    }
    v16h a0 = frag_a_pitch(ldsA[cur] + (wm * 32) * PITCH, PITCH);
    v16h a1 = frag_a_pitch(ldsA[cur] + (wm * 32 + 16) * PITCH, PITCH);
    v16h b0 = frag_b_pitch(ldsB[cur] + (wn * 32) * PITCH, PITCH);
    v16h b1 = frag_b_pitch(ldsB[cur] + (wn * 32 + 16) * PITCH, PITCH);
    acc[0][0] = wmma32(a0, b0, acc[0][0]);
    acc[0][1] = wmma32(a0, b1, acc[0][1]);
    acc[1][0] = wmma32(a1, b0, acc[1][0]);
    acc[1][1] = wmma32(a1, b1, acc[1][1]);
    stage_fence();
  }
  int lane = threadIdx.x & 31, nloc = lane & 15, rbase = (lane >> 4) * 8;
#pragma unroll
  for (int mf = 0; mf < 2; ++mf)
#pragma unroll
    for (int nf = 0; nf < 2; ++nf)
#pragma unroll
      for (int i = 0; i < 8; ++i) {
        int row = m0 + wm * 32 + mf * 16 + rbase + i;
        int c = n0 + wn * 32 + nf * 16 + nloc;
        relu[(size_t)row * 512 + c] = (half_t)fmaxf(acc[mf][nf][i], 0.f);
      }
}

extern "C" void kernel_launch(void* const* d_in, const int* in_sizes, int n_in,
                              void* d_out, int out_size, void* d_ws, size_t ws_size,
                              hipStream_t stream) {
  const float* x    = (const float*)d_in[0];
  const float* src  = (const float*)d_in[1];
  const float* xpe  = (const float*)d_in[2];
  const float* spe  = (const float*)d_in[3];
  const float* Wq   = (const float*)d_in[4];
  const float* Wk   = (const float*)d_in[5];
  const float* Wv   = (const float*)d_in[6];
  const float* Wm   = (const float*)d_in[7];
  const float* W1   = (const float*)d_in[8];
  const float* W2   = (const float*)d_in[9];
  const float* g1   = (const float*)d_in[10];
  const float* b1   = (const float*)d_in[11];
  const float* g2   = (const float*)d_in[12];
  const float* b2   = (const float*)d_in[13];
  float* out = (float*)d_out;

  char* w = (char*)d_ws;
  auto carve = [&](size_t bytes) {
    char* p = w;
    w += (bytes + 255) & ~(size_t)255;
    return p;
  };
  half_t* xh    = (half_t*)carve((size_t)MR * CC * 2);
  half_t* sh    = (half_t*)carve((size_t)MR * CC * 2);
  half_t* Wqh   = (half_t*)carve((size_t)CC * CC * 2);
  half_t* Wkh   = (half_t*)carve((size_t)CC * CC * 2);
  half_t* Wvh   = (half_t*)carve((size_t)CC * CC * 2);
  half_t* Wmh   = (half_t*)carve((size_t)CC * CC * 2);
  half_t* W1h   = (half_t*)carve((size_t)512 * 512 * 2);
  half_t* W2h   = (half_t*)carve((size_t)256 * 512 * 2);
  half_t* Qphi  = (half_t*)carve((size_t)4 * HH * Lp * DD * 2);
  half_t* KphiT = (half_t*)carve((size_t)4 * HH * DD * Lq * 2);
  half_t* VscT  = (half_t*)carve((size_t)4 * HH * DD * Lq * 2);
  float*  Ksum  = (float*)carve((size_t)4 * HH * DD * 4);
  float*  KV    = (float*)carve((size_t)4 * HH * DD * DD * 4);
  half_t* msgh  = (half_t*)carve((size_t)MR * CC * 2);
  half_t* cath  = (half_t*)carve((size_t)MR * 512 * 2);
  half_t* reluh = (half_t*)carve((size_t)MR * 512 * 2);

  auto cvt = [&](const float* in, half_t* o, int n) {
    cvt_f16<<<(n + 255) / 256, 256, 0, stream>>>(in, o, n);
  };
  cvt(x,  xh,  MR * CC);
  cvt(src, sh, MR * CC);
  cvt(Wq, Wqh, CC * CC);
  cvt(Wk, Wkh, CC * CC);
  cvt(Wv, Wvh, CC * CC);
  cvt(Wm, Wmh, CC * CC);
  cvt(W1, W1h, 512 * 512);
  cvt(W2, W2h, 256 * 512);

  dim3 pg(MR / 128, CC / 64);
  proj_kernel<0><<<pg, 256, 0, stream>>>(xh, Wqh, xpe, Qphi);
  proj_kernel<1><<<pg, 256, 0, stream>>>(sh, Wkh, spe, KphiT);
  proj_kernel<2><<<pg, 256, 0, stream>>>(sh, Wvh, nullptr, VscT);

  ksum_kernel<<<4 * HH * DD, 256, 0, stream>>>(KphiT, Ksum);
  kv_kernel<<<4 * HH, 256, 0, stream>>>(KphiT, VscT, KV);
  msg_kernel<<<dim3(Lp / 256, 4 * HH), 256, 0, stream>>>(Qphi, KV, Ksum, msgh);

  gemm_ln_kernel<0, 256><<<MR / 64, 256, 0, stream>>>(msgh, Wmh, xh, x, g1, b1, cath, nullptr);
  mlp1_kernel<<<dim3(MR / 128, 512 / 64), 256, 0, stream>>>(cath, W1h, reluh);
  gemm_ln_kernel<1, 512><<<MR / 64, 256, 0, stream>>>(reluh, W2h, nullptr, x, g2, b2, nullptr, out);
}
